// AttLayer_17265768529961
// MI455X (gfx1250) — matrix operand order repaired and checked by reading, not compiled, not run on hardware
//
#include <hip/hip_runtime.h>
#include <hip/hip_bf16.h>

#define L_TOT 131072
#define C_IN  256
#define CH    128
#define BLK   512
#define HALFW 256
#define WINW  1024
#define NB    (L_TOT / BLK)
#define VS    (L_TOT + 2 * HALFW)   // padded sequence length for k/v

typedef __attribute__((ext_vector_type(16))) _Float16 v16h;
typedef __attribute__((ext_vector_type(8)))  _Float16 v8h;
typedef __attribute__((ext_vector_type(8)))  float    v8f;

// A-matrix fragment (16x32 f16): lane&15 = M row; lanes<16 hold K 0..7,16..23,
// lanes>=16 hold K 8..15,24..31 (CDNA5 ISA 7.12.2).
__device__ __forceinline__ v16h load_fragA(const _Float16* base, int stride, int lane) {
  const _Float16* p = base + (lane & 15) * stride + ((lane >> 4) << 3);
  v8h lo = *(const v8h*)(p);
  v8h hi = *(const v8h*)(p + 16);
  return __builtin_shufflevector(lo, hi, 0,1,2,3,4,5,6,7,8,9,10,11,12,13,14,15);
}

// B-matrix fragment (32x16 f16), source stored as [N][K] (i.e. transposed):
// lane&15 = N col; lanes<16 hold K 0..7,16..23, lanes>=16 hold K 8..15,24..31.
__device__ __forceinline__ v16h load_fragB(const _Float16* base, int stride, int lane) {
  const _Float16* p = base + (lane & 15) * stride + ((lane >> 4) << 3);
  v8h lo = *(const v8h*)(p);
  v8h hi = *(const v8h*)(p + 16);
  return __builtin_shufflevector(lo, hi, 0,1,2,3,4,5,6,7,8,9,10,11,12,13,14,15);
}

__device__ __forceinline__ v8f wmma16(v16h a, v16h b, v8f c) {
  return __builtin_amdgcn_wmma_f32_16x16x32_f16(false, a, false, b, (short)0, c, false, false);
}

// ---------------- prep: weight f32->f16 conversion + zero-pad k/v edges ----
__global__ __launch_bounds__(256) void k_prep(const float* __restrict__ Wq,
                                              const float* __restrict__ Wk,
                                              const float* __restrict__ Wv,
                                              const float* __restrict__ Wo,
                                              _Float16* __restrict__ Wq16,
                                              _Float16* __restrict__ Wk16,
                                              _Float16* __restrict__ Wv16,
                                              _Float16* __restrict__ Wo16,
                                              _Float16* __restrict__ khT,
                                              _Float16* __restrict__ vh) {
  int id = blockIdx.x * 256 + threadIdx.x;
  if (id < 32768) {
    Wq16[id] = (_Float16)Wq[id];
  } else if (id < 65536) {
    int j = id - 32768;  Wk16[j] = (_Float16)Wk[j];
  } else if (id < 98304) {
    int j = id - 65536;  Wv16[j] = (_Float16)Wv[j];
  } else if (id < 131072) {
    int j = id - 98304;  Wo16[j] = (_Float16)Wo[j];
  } else if (id < 196608) {              // khT pad rows [0,256) and [L+256, L+512)
    int j = id - 131072;
    int row = j >> 7, col = j & 127;
    int pr = (row < HALFW) ? row : (L_TOT + row);
    khT[(size_t)pr * CH + col] = (_Float16)0.f;
  } else if (id < 262144) {              // vh pad cols
    int j = id - 196608;
    int c = j >> 9, col = j & 511;
    int pl = (col < HALFW) ? col : (L_TOT + col);
    vh[(size_t)c * VS + pl] = (_Float16)0.f;
  }
}

// ---------------- QKV projection: [128x256] x [256 x Ltile] ----------------
__global__ __launch_bounds__(256) void k_qkv(const float* __restrict__ x1,
                                             const _Float16* __restrict__ Wq16,
                                             const _Float16* __restrict__ Wk16,
                                             const _Float16* __restrict__ Wv16,
                                             const float* __restrict__ bq,
                                             const float* __restrict__ bk,
                                             const float* __restrict__ bv,
                                             _Float16* __restrict__ qh,
                                             _Float16* __restrict__ khT,
                                             _Float16* __restrict__ vh) {
  __shared__ _Float16 xT[128 * 256];     // x1 tile, transposed [l][c], 64KB
  __shared__ _Float16 vstage[128 * 128]; // v tile staging [co][l], 32KB
  const int tid = threadIdx.x;
  const int lane = tid & 31, wave = tid >> 5;
  const int l0 = blockIdx.x * 128;

  __builtin_prefetch(&x1[(size_t)(tid) * L_TOT + l0], 0, 0);
  for (int it = 0; it < 128; ++it) {
    int idx = it * 256 + tid;
    int c = idx >> 7, lc = idx & 127;
    xT[lc * 256 + c] = (_Float16)x1[(size_t)c * L_TOT + l0 + lc];
  }
  __syncthreads();

  const int co0 = wave * 16;
  v16h aQ[8], aK[8], aV[8];
#pragma unroll
  for (int ks = 0; ks < 8; ++ks) {
    aQ[ks] = load_fragA(Wq16 + co0 * C_IN + ks * 32, C_IN, lane);
    aK[ks] = load_fragA(Wk16 + co0 * C_IN + ks * 32, C_IN, lane);
    aV[ks] = load_fragA(Wv16 + co0 * C_IN + ks * 32, C_IN, lane);
  }
  const int mrow = (lane >> 4) << 3;
  float bQ[8], bK[8], bV[8];
#pragma unroll
  for (int e = 0; e < 8; ++e) {
    bQ[e] = bq[co0 + mrow + e];
    bK[e] = bk[co0 + mrow + e];
    bV[e] = bv[co0 + mrow + e];
  }

  for (int nt = 0; nt < 8; ++nt) {
    v16h bX[8];
#pragma unroll
    for (int ks = 0; ks < 8; ++ks)
      bX[ks] = load_fragB(&xT[(nt * 16) * 256 + ks * 32], 256, lane);
    v8f accQ, accK, accV;
#pragma unroll
    for (int e = 0; e < 8; ++e) { accQ[e] = 0.f; accK[e] = 0.f; accV[e] = 0.f; }
#pragma unroll
    for (int ks = 0; ks < 8; ++ks) {
      accQ = wmma16(aQ[ks], bX[ks], accQ);
      accK = wmma16(aK[ks], bX[ks], accK);
      accV = wmma16(aV[ks], bX[ks], accV);
    }
    const int colin = nt * 16 + (lane & 15);
    const int col = l0 + colin;
    // q/k: element e -> consecutive channel co; pack into one b128 store each
    v8h hq, hk;
#pragma unroll
    for (int e = 0; e < 8; ++e) {
      hq[e] = (_Float16)(accQ[e] + bQ[e]);
      hk[e] = (_Float16)(accK[e] + bK[e]);
    }
    *(v8h*)&qh[(size_t)col * CH + co0 + mrow]            = hq;
    *(v8h*)&khT[(size_t)(HALFW + col) * CH + co0 + mrow] = hk;
    // v: strided by VS per element -> stage in LDS, coalesced writeout below
#pragma unroll
    for (int e = 0; e < 8; ++e)
      vstage[(co0 + mrow + e) * 128 + colin] = (_Float16)(accV[e] + bV[e]);
  }
  __syncthreads();
  // coalesced v writeout: 128x128 halves, 8 x b128 per thread
#pragma unroll
  for (int i = 0; i < 8; ++i) {
    int idx = (i * 256 + tid) * 8;
    int co = idx >> 7, cc = idx & 127;
    *(v8h*)&vh[(size_t)co * VS + HALFW + l0 + cc] = *(const v8h*)&vstage[co * 128 + cc];
  }
}

// ---------------- windowed attention, flash-style online softmax -----------
__global__ __launch_bounds__(256) void k_attn(const _Float16* __restrict__ qh,
                                              const _Float16* __restrict__ khT,
                                              const _Float16* __restrict__ vh,
                                              const float* __restrict__ mask,
                                              _Float16* __restrict__ aout) {
  __shared__ float    fmld[WINW];          // 4KB: window mask (fm)
  __shared__ _Float16 Pld[8 * 16 * 128];   // 32KB: per-wave P / O staging [m][w]
  const int tid = threadIdx.x, lane = tid & 31, wave = tid >> 5;
  const int n = blockIdx.x;
  const int lblk = n * BLK;

  for (int w = tid; w < WINW; w += 256) {
    int g = lblk - HALFW + w;
    float m = (g >= 0 && g < L_TOT) ? mask[g] : 0.f;
    fmld[w] = (w < 2 * BLK - 1) ? m : 0.f;
  }
  __syncthreads();

  const float scale = 0.08838834764831845f;  // 1/sqrt(128)
  _Float16* myP = &Pld[wave * 2048];
  const int mrow = (lane >> 4) << 3;

  for (int mt = 0; mt < 4; ++mt) {
    const int q0 = mt * 128 + wave * 16;
    v16h aq[4];
#pragma unroll
    for (int ks = 0; ks < 4; ++ks)
      aq[ks] = load_fragA(qh + (size_t)(lblk + q0) * CH + ks * 32, CH, lane);

    float rm[8], rs[8];
    v8f Oacc[8];
#pragma unroll
    for (int e = 0; e < 8; ++e) { rm[e] = -1e30f; rs[e] = 0.f; }
#pragma unroll
    for (int t = 0; t < 8; ++t)
#pragma unroll
      for (int e = 0; e < 8; ++e) Oacc[t][e] = 0.f;

    for (int ch = 0; ch < 8; ++ch) {
      const int wbase = ch * 128;
      v8f S[8];
#pragma unroll
      for (int nt = 0; nt < 8; ++nt) {
        v8f acc;
#pragma unroll
        for (int e = 0; e < 8; ++e) acc[e] = 0.f;
        const _Float16* kb = khT + (size_t)(lblk + wbase + nt * 16) * CH;
#pragma unroll
        for (int ks = 0; ks < 4; ++ks)
          acc = wmma16(aq[ks], load_fragB(kb + ks * 32, CH, lane), acc);
        float lg = __logf(fmld[wbase + nt * 16 + (lane & 15)] + 1e-6f);
#pragma unroll
        for (int e = 0; e < 8; ++e) S[nt][e] = acc[e] * scale + lg;
      }
      // chunk row-max (reduce across 16-lane halves: matches C-layout rows)
      float sc[8];
#pragma unroll
      for (int e = 0; e < 8; ++e) {
        float m = S[0][e];
#pragma unroll
        for (int nt = 1; nt < 8; ++nt) m = fmaxf(m, S[nt][e]);
#pragma unroll
        for (int off = 1; off < 16; off <<= 1) m = fmaxf(m, __shfl_xor(m, off, 32));
        float nm = fmaxf(rm[e], m);
        sc[e] = __expf(rm[e] - nm);
        rm[e] = nm;
        rs[e] *= sc[e];
      }
#pragma unroll
      for (int t = 0; t < 8; ++t)
#pragma unroll
        for (int e = 0; e < 8; ++e) Oacc[t][e] *= sc[e];
      // P = exp(S - max), stage to LDS re-laid out as A-matrix [m][w]
#pragma unroll
      for (int nt = 0; nt < 8; ++nt) {
        int wc = nt * 16 + (lane & 15);
        float fm = fmld[wbase + wc];
#pragma unroll
        for (int e = 0; e < 8; ++e) {
          float p = __expf(S[nt][e] - rm[e]);
          rs[e] += p;
          myP[(mrow + e) * 128 + wc] = (_Float16)(p * fm);
        }
      }
      asm volatile("s_wait_dscnt 0" ::: "memory");
      // O += P * V^T
#pragma unroll
      for (int ks2 = 0; ks2 < 4; ++ks2) {
        v16h aP = load_fragA(myP + ks2 * 32, 128, lane);
#pragma unroll
        for (int t = 0; t < 8; ++t) {
          v16h bV = load_fragB(vh + (size_t)(t * 16) * VS + (lblk + wbase + ks2 * 32),
                               VS, lane);
          Oacc[t] = wmma16(aP, bV, Oacc[t]);
        }
      }
      asm volatile("s_wait_dscnt 0" ::: "memory");
    }
    // finalize: 1/rowsum, relu, stage O tile in LDS, coalesced f16 writeout
    float inv[8];
#pragma unroll
    for (int e = 0; e < 8; ++e) {
      float s = rs[e];
#pragma unroll
      for (int off = 1; off < 16; off <<= 1) s += __shfl_xor(s, off, 32);
      inv[e] = 1.f / s;
    }
#pragma unroll
    for (int t = 0; t < 8; ++t) {
      int cv = t * 16 + (lane & 15);
#pragma unroll
      for (int e = 0; e < 8; ++e)
        myP[(mrow + e) * 128 + cv] = (_Float16)fmaxf(Oacc[t][e] * inv[e], 0.f);
    }
    asm volatile("s_wait_dscnt 0" ::: "memory");
    {
      int row = lane >> 1, base = (lane & 1) * 64;
#pragma unroll
      for (int i = 0; i < 8; ++i)
        *(v8h*)&aout[(size_t)(lblk + q0 + row) * CH + base + i * 8] =
            *(const v8h*)&myP[row * 128 + base + i * 8];
    }
    asm volatile("s_wait_dscnt 0" ::: "memory");
  }
}

// ---------------- output projection: [256x128] x [128 x Ltile] -------------
__global__ __launch_bounds__(256) void k_out(const _Float16* __restrict__ aout,
                                             const _Float16* __restrict__ Wo16,
                                             const float* __restrict__ bo,
                                             const float* __restrict__ mask,
                                             float* __restrict__ out) {
  const int tid = threadIdx.x, lane = tid & 31, wave = tid >> 5;
  const int l0 = blockIdx.x * 128;
  const int mrow = (lane >> 4) << 3;
  for (int mt = 0; mt < 2; ++mt) {
    const int o0 = wave * 32 + mt * 16;
    v16h aW[4];
#pragma unroll
    for (int ks = 0; ks < 4; ++ks)
      aW[ks] = load_fragA(Wo16 + o0 * CH + ks * 32, CH, lane);
    float bov[8];
#pragma unroll
    for (int e = 0; e < 8; ++e) bov[e] = bo[o0 + mrow + e];
    for (int nt = 0; nt < 8; ++nt) {
      v8f acc;
#pragma unroll
      for (int e = 0; e < 8; ++e) acc[e] = 0.f;
#pragma unroll
      for (int ks = 0; ks < 4; ++ks) {
        v16h bA = load_fragB(aout + (size_t)(l0 + nt * 16) * CH + ks * 32, CH, lane);
        acc = wmma16(aW[ks], bA, acc);
      }
      int nl = l0 + nt * 16 + (lane & 15);
      float mv = mask[nl];
#pragma unroll
      for (int e = 0; e < 8; ++e)
        out[(size_t)(o0 + mrow + e) * L_TOT + nl] = (acc[e] + bov[e]) * mv;
    }
  }
}

extern "C" void kernel_launch(void* const* d_in, const int* in_sizes, int n_in,
                              void* d_out, int out_size, void* d_ws, size_t ws_size,
                              hipStream_t stream) {
  (void)in_sizes; (void)n_in; (void)out_size; (void)ws_size;
  const float* x1   = (const float*)d_in[0];
  const float* mask = (const float*)d_in[1];
  const float* Wq   = (const float*)d_in[2];
  const float* bq   = (const float*)d_in[3];
  const float* Wk   = (const float*)d_in[4];
  const float* bk   = (const float*)d_in[5];
  const float* Wv   = (const float*)d_in[6];
  const float* bv   = (const float*)d_in[7];
  const float* Wo   = (const float*)d_in[8];
  const float* bo   = (const float*)d_in[9];
  float* out = (float*)d_out;

  _Float16* w = (_Float16*)d_ws;
  _Float16* Wq16 = w;  w += 32768;
  _Float16* Wk16 = w;  w += 32768;
  _Float16* Wv16 = w;  w += 32768;
  _Float16* Wo16 = w;  w += 32768;
  _Float16* qh   = w;  w += (size_t)L_TOT * CH;
  _Float16* khT  = w;  w += (size_t)VS * CH;
  _Float16* vh   = w;  w += (size_t)CH * VS;
  _Float16* aout = w;

  k_prep<<<1024, 256, 0, stream>>>(Wq, Wk, Wv, Wo, Wq16, Wk16, Wv16, Wo16, khT, vh);
  k_qkv<<<L_TOT / 128, 256, 0, stream>>>(x1, Wq16, Wk16, Wv16, bq, bk, bv, qh, khT, vh);
  k_attn<<<NB, 256, 0, stream>>>(qh, khT, vh, mask, aout);
  k_out<<<L_TOT / 128, 256, 0, stream>>>(aout, Wo16, bo, mask, out);
}
